// DeepMIL_TFL_MoE_86741159510167
// MI455X (gfx1250) — matrix operand order repaired and checked by reading, not compiled, not run on hardware
//
#include <hip/hip_runtime.h>
#include <hip/hip_bf16.h>

#define DIM_IN   768
#define DIM_EMB  512
#define DIM_ATTN 384
#define N_EXPERTS 13
#define TOPK      5
#define NUM_CLS   2
#define K_INST 100000
#define TILE_M 128
#define NTILES ((K_INST + TILE_M - 1) / TILE_M)   // 782

typedef __attribute__((ext_vector_type(16))) __bf16 v16bf;
typedef __attribute__((ext_vector_type(8)))  __bf16 v8bf;
typedef __attribute__((ext_vector_type(8)))  float  v8f;

// ---------------------------------------------------------------------------
// Kernel 0: fp32 -> bf16 weight conversion + transpose to [N][K] layout so a
// WMMA B-fragment is one contiguous 32B load per lane.
// ---------------------------------------------------------------------------
__global__ void prep_weights(const float* __restrict__ Wp,
                             const float* __restrict__ Wa1,
                             __bf16* __restrict__ WpT,
                             __bf16* __restrict__ Wa1T) {
  int i = blockIdx.x * blockDim.x + threadIdx.x;
  const int total1 = DIM_EMB * DIM_IN;          // WpT: [512][768]
  const int total2 = DIM_ATTN * DIM_EMB;        // Wa1T: [384][512]
  if (i < total1) {
    int n = i / DIM_IN, c = i % DIM_IN;
    WpT[i] = (__bf16)Wp[c * DIM_EMB + n];       // W_proj is [768][512]
  } else if (i < total1 + total2) {
    int j = i - total1;
    int n = j / DIM_EMB, c = j % DIM_EMB;
    Wa1T[j] = (__bf16)Wa1[c * DIM_ATTN + n];    // W_attn1 is [512][384]
  }
}

// ---------------------------------------------------------------------------
// Kernel 1: fused  H = relu(X*Wp+b) ; T = tanh(H*Wa1+b1) ; s = T*w2+b2 ;
// per-tile online-softmax partials (max, sumexp, exp-weighted H sum).
// 512 threads = 16 waves; wave (rowTile 0..7, colHalf 0..1).
// ---------------------------------------------------------------------------
__launch_bounds__(512, 1)
__global__ void mil_tile_kernel(const float* __restrict__ X,
                                const __bf16* __restrict__ WpT,
                                const __bf16* __restrict__ Wa1T,
                                const float* __restrict__ b_proj,
                                const float* __restrict__ b_attn1,
                                const float* __restrict__ W_attn2,
                                const float* __restrict__ b_attn2,
                                float* __restrict__ mBuf,
                                float* __restrict__ zBuf,
                                float* __restrict__ pBuf) {
  __shared__ __bf16 Hs[TILE_M * DIM_EMB];   // 128 KB bf16 tile of H
  __shared__ float  sScore[TILE_M];

  const int tid     = threadIdx.x;
  const int wave    = tid >> 5;
  const int lane    = tid & 31;
  const int rowTile = wave >> 1;            // 0..7  (16 rows each)
  const int colHalf = wave & 1;             // 0..1  (256 cols each)
  const int half    = lane >> 4;            // K-split half per WMMA layout
  const int ln16    = lane & 15;
  const int blockRow0 = blockIdx.x * TILE_M;

  if (tid < TILE_M) {
    sScore[tid] = (blockRow0 + tid < K_INST) ? b_attn2[0] : -3.0e38f;
  }

  // ---------------- GEMM1: H = relu(X * Wp + b_proj) ----------------
  v8f zero = {0.f, 0.f, 0.f, 0.f, 0.f, 0.f, 0.f, 0.f};
  v8f acc[16];
#pragma unroll
  for (int t = 0; t < 16; ++t) acc[t] = zero;

  int grow = blockRow0 + rowTile * 16 + ln16;
  if (grow >= K_INST) grow = K_INST - 1;            // clamp; masked by score
  const float* xrow = X + (long)grow * DIM_IN;

#pragma unroll 1
  for (int kb = 0; kb < DIM_IN; kb += 32) {
    int k0 = kb + half * 16;
    v8f xa = *(const v8f*)(xrow + k0);
    v8f xb = *(const v8f*)(xrow + k0 + 8);
    __builtin_prefetch(xrow + kb + 32, 0, 1);       // global_prefetch_b8
    v16bf afrag;
#pragma unroll
    for (int e = 0; e < 8; ++e) { afrag[e] = (__bf16)xa[e]; afrag[8 + e] = (__bf16)xb[e]; }
#pragma unroll
    for (int t = 0; t < 16; ++t) {
      const __bf16* bp = WpT + (long)(colHalf * 256 + t * 16 + ln16) * DIM_IN
                             + kb + half * 16;
      v16bf bfrag = *(const v16bf*)bp;
      acc[t] = __builtin_amdgcn_wmma_f32_16x16x32_bf16(
                 false, afrag, false, bfrag, (short)0, acc[t], false, false);
    }
  }

  // bias + ReLU, store bf16 H tile to LDS (C layout: col per lane, row per vgpr)
#pragma unroll
  for (int t = 0; t < 16; ++t) {
    int col = colHalf * 256 + t * 16 + ln16;
    float bv = b_proj[col];
#pragma unroll
    for (int v = 0; v < 8; ++v) {
      int row = rowTile * 16 + v + half * 8;
      float h = acc[t][v] + bv;
      h = h > 0.f ? h : 0.f;
      Hs[row * DIM_EMB + col] = (__bf16)h;
    }
  }
  __syncthreads();

  // ---------------- GEMM2: T = tanh(H * Wa1 + b1);  s = T * w2 ----------------
  v8f acc2[12];
#pragma unroll
  for (int t = 0; t < 12; ++t) acc2[t] = zero;

  {
    const __bf16* hrow = Hs + (rowTile * 16 + ln16) * DIM_EMB;
#pragma unroll 1
    for (int kb = 0; kb < DIM_EMB; kb += 32) {
      int k0 = kb + half * 16;
      v8bf lo = *(const v8bf*)(hrow + k0);
      v8bf hi = *(const v8bf*)(hrow + k0 + 8);
      v16bf afrag;
#pragma unroll
      for (int e = 0; e < 8; ++e) { afrag[e] = lo[e]; afrag[8 + e] = hi[e]; }
#pragma unroll
      for (int t = 0; t < 12; ++t) {
        const __bf16* bp = Wa1T + (long)(colHalf * 192 + t * 16 + ln16) * DIM_EMB
                                + kb + half * 16;
        v16bf bfrag = *(const v16bf*)bp;
        acc2[t] = __builtin_amdgcn_wmma_f32_16x16x32_bf16(
                    false, afrag, false, bfrag, (short)0, acc2[t], false, false);
      }
    }
  }

  // per-lane partial score, reduce over the 16 lanes that share a row
  float part[8];
#pragma unroll
  for (int v = 0; v < 8; ++v) part[v] = 0.f;
#pragma unroll
  for (int t = 0; t < 12; ++t) {
    int n = colHalf * 192 + t * 16 + ln16;
    float bv = b_attn1[n];
    float w2 = W_attn2[n];
#pragma unroll
    for (int v = 0; v < 8; ++v) part[v] += tanhf(acc2[t][v] + bv) * w2;
  }
#pragma unroll
  for (int v = 0; v < 8; ++v) {
#pragma unroll
    for (int mm = 8; mm >= 1; mm >>= 1) part[v] += __shfl_xor(part[v], mm, 16);
    if (ln16 == 0)
      atomicAdd(&sScore[rowTile * 16 + v + half * 8], part[v]);  // ds_add_f32
  }
  __syncthreads();

  // ---------------- online softmax partials + weighted H pooling ----------------
  float mmax = -3.0e38f;
  for (int k = 0; k < TILE_M; ++k) mmax = fmaxf(mmax, sScore[k]);
  __syncthreads();
  if (tid < TILE_M) sScore[tid] = __expf(sScore[tid] - mmax);
  __syncthreads();

  {                                           // one column per thread (512 cols)
    int c = tid;
    float p = 0.f;
    for (int k = 0; k < TILE_M; ++k) p += sScore[k] * (float)Hs[k * DIM_EMB + c];
    pBuf[(long)blockIdx.x * DIM_EMB + c] = p;
  }
  if (tid == 0) {
    float z = 0.f;
    for (int k = 0; k < TILE_M; ++k) z += sScore[k];
    mBuf[blockIdx.x] = mmax;
    zBuf[blockIdx.x] = z;
  }
}

// ---------------------------------------------------------------------------
// Kernel 2: cross-tile softmax reduction, router, top-5 MoE, pred head.
// ---------------------------------------------------------------------------
__launch_bounds__(256, 1)
__global__ void finalize_kernel(const float* __restrict__ mBuf,
                                const float* __restrict__ zBuf,
                                const float* __restrict__ pBuf,
                                const float* __restrict__ W_router,
                                const float* __restrict__ X_tfl,
                                const float* __restrict__ W_pred,
                                const float* __restrict__ b_pred,
                                float* __restrict__ out) {
  __shared__ float red[256];
  __shared__ float Mvec[DIM_EMB];
  __shared__ float rl[16];
  __shared__ float w5[TOPK];
  __shared__ int   i5[TOPK];
  const int tid = threadIdx.x;

  // global max of per-tile maxima
  float lm = -3.0e38f;
  for (int i = tid; i < NTILES; i += 256) lm = fmaxf(lm, mBuf[i]);
  red[tid] = lm;
  __syncthreads();
  for (int s = 128; s > 0; s >>= 1) {
    if (tid < s) red[tid] = fmaxf(red[tid], red[tid + s]);
    __syncthreads();
  }
  float Mstar = red[0];
  __syncthreads();

  // global sum of exp (rescaled)
  float ls = 0.f;
  for (int i = tid; i < NTILES; i += 256) ls += zBuf[i] * __expf(mBuf[i] - Mstar);
  red[tid] = ls;
  __syncthreads();
  for (int s = 128; s > 0; s >>= 1) {
    if (tid < s) red[tid] += red[tid + s];
    __syncthreads();
  }
  float S = red[0];
  __syncthreads();

  // slide representation M
  for (int c = tid; c < DIM_EMB; c += 256) {
    float a = 0.f;
    for (int i = 0; i < NTILES; ++i)
      a += pBuf[(long)i * DIM_EMB + c] * __expf(mBuf[i] - Mstar);
    Mvec[c] = a / S;
  }
  __syncthreads();

  // router logits
  if (tid < N_EXPERTS) {
    float a = 0.f;
    for (int c = 0; c < DIM_EMB; ++c) a += Mvec[c] * W_router[c * N_EXPERTS + tid];
    rl[tid] = a;
  }
  __syncthreads();

  // softmax + top-5 (tiny, single thread)
  if (tid == 0) {
    float mx = rl[0];
    for (int j = 1; j < N_EXPERTS; ++j) mx = fmaxf(mx, rl[j]);
    float probs[N_EXPERTS]; float s = 0.f;
    for (int j = 0; j < N_EXPERTS; ++j) { probs[j] = __expf(rl[j] - mx); s += probs[j]; }
    for (int j = 0; j < N_EXPERTS; ++j) probs[j] /= s;
    for (int i = 0; i < TOPK; ++i) {
      int bi = 0; float bv = -1.f;
      for (int j = 0; j < N_EXPERTS; ++j) if (probs[j] > bv) { bv = probs[j]; bi = j; }
      w5[i] = bv; i5[i] = bi; probs[bi] = -2.f;
    }
  }
  __syncthreads();

  // MoE combine of identity experts (reuse Mvec as moe_output)
  for (int c = tid; c < DIM_EMB; c += 256) {
    float a = 0.f;
    for (int i = 0; i < TOPK; ++i) a += w5[i] * X_tfl[i5[i] * DIM_EMB + c];
    Mvec[c] = a;
  }
  __syncthreads();

  if (tid < NUM_CLS) {
    float a = b_pred[tid];
    for (int c = 0; c < DIM_EMB; ++c) a += Mvec[c] * W_pred[c * NUM_CLS + tid];
    out[tid] = a;
  }
}

// ---------------------------------------------------------------------------
extern "C" void kernel_launch(void* const* d_in, const int* in_sizes, int n_in,
                              void* d_out, int out_size, void* d_ws, size_t ws_size,
                              hipStream_t stream) {
  const float* X_tfl    = (const float*)d_in[0];
  const float* X        = (const float*)d_in[1];
  const float* W_proj   = (const float*)d_in[2];
  const float* b_proj   = (const float*)d_in[3];
  const float* W_attn1  = (const float*)d_in[4];
  const float* b_attn1  = (const float*)d_in[5];
  const float* W_attn2  = (const float*)d_in[6];
  const float* b_attn2  = (const float*)d_in[7];
  const float* W_router = (const float*)d_in[8];
  const float* W_pred   = (const float*)d_in[9];
  const float* b_pred   = (const float*)d_in[10];
  float* out = (float*)d_out;

  char* ws = (char*)d_ws;
  __bf16* WpT  = (__bf16*)(ws);                 // 512*768*2  = 786432 B
  __bf16* Wa1T = (__bf16*)(ws + 786432);        // 384*512*2  = 393216 B
  float*  mBuf = (float*)(ws + 1179648);        // NTILES (padded 1024) f32
  float*  zBuf = (float*)(ws + 1183744);        // NTILES (padded 1024) f32
  float*  pBuf = (float*)(ws + 1187840);        // NTILES*512 f32

  const int prepTotal = DIM_EMB * DIM_IN + DIM_ATTN * DIM_EMB;
  prep_weights<<<(prepTotal + 255) / 256, 256, 0, stream>>>(W_proj, W_attn1, WpT, Wa1T);
  mil_tile_kernel<<<NTILES, 512, 0, stream>>>(X, WpT, Wa1T, b_proj, b_attn1,
                                              W_attn2, b_attn2, mBuf, zBuf, pBuf);
  finalize_kernel<<<1, 256, 0, stream>>>(mBuf, zBuf, pBuf, W_router, X_tfl,
                                         W_pred, b_pred, out);
}
